// Reformer_85521388798118
// MI455X (gfx1250) — compile-verified, matrix operand order checked by reading, not run on hardware
//
#include <hip/hip_runtime.h>
#include <hip/hip_bf16.h>
#include <stdint.h>

// ---------------------------------------------------------------------------
// Types for CDNA5 WMMA (wave32): v16bf A/B fragments, v8f accumulator.
// ---------------------------------------------------------------------------
typedef __bf16 bf16_t;
typedef bf16_t v16bf __attribute__((ext_vector_type(16)));
typedef bf16_t v8bf  __attribute__((ext_vector_type(8)));
typedef float  v8f   __attribute__((ext_vector_type(8)));

#define DEVINL __device__ __forceinline__

static constexpr int kB   = 4;
static constexpr int kS   = 4096;
static constexpr int kD   = 512;
static constexpr int kH   = 8;
static constexpr int kNB  = 64;
static constexpr int kR   = 2;
static constexpr int kDH  = 64;   // D / H
static constexpr int kDFF = 2048;
static constexpr int kM   = kB * kS;        // 16384 rows
static constexpr int kRBH = kR * kB * kH;   // 64

DEVINL bf16_t f2bf(float f) {
  uint32_t u = __builtin_bit_cast(uint32_t, f);
  uint32_t r = u + 0x7FFFu + ((u >> 16) & 1u);   // round-to-nearest-even
  return __builtin_bit_cast(bf16_t, (uint16_t)(r >> 16));
}

// ---------------------------------------------------------------------------
// CDNA5 async global->LDS copy (GLOBAL_LOAD_ASYNC_TO_LDS_B128, ASYNCcnt).
// LDS byte address = low 32 bits of the generic shared pointer (ISA 10.2:
// LDS aperture maps addr[31:0] to the wave-relative LDS offset).
// ---------------------------------------------------------------------------
DEVINL void async_copy_b128(const void* lds_dst, const void* gsrc) {
  const uint32_t laddr = (uint32_t)(uintptr_t)lds_dst;
  asm volatile("global_load_async_to_lds_b128 %0, %1, off"
               :: "v"(laddr), "v"(gsrc)
               : "memory");
}

template <int N> DEVINL void wait_asynccnt() {
#if __has_builtin(__builtin_amdgcn_s_wait_asynccnt)
  __builtin_amdgcn_s_wait_asynccnt(N);
#else
  asm volatile("s_wait_asynccnt %0" :: "n"(N) : "memory");
#endif
}

// Fragment loader for 16-bit A/B operands of V_WMMA_F32_16X16X32_BF16.
// Per ISA 7.12.2: lane L (0..15) holds row L, K = {0..7, 16..23};
// lane L+16 holds row L, K = {8..15, 24..31}. Both operands are fetched as
// contiguous rows (B is kept pre-transposed so its "row" is a column of B).
DEVINL v16bf load_frag16(const bf16_t* base, int stride, int lane) {
  const bf16_t* p = base + (lane & 15) * stride + ((lane >> 4) & 1) * 8;
  v8bf lo = *(const v8bf*)p;
  v8bf hi = *(const v8bf*)(p + 16);
  return __builtin_shufflevector(lo, hi, 0,1,2,3,4,5,6,7,8,9,10,11,12,13,14,15);
}

DEVINL v8f wmma_bf16(v16bf a, v16bf b, v8f c) {
  return __builtin_amdgcn_wmma_f32_16x16x32_bf16(false, a, false, b, (short)0, c,
                                                 false, false);
}

// ---------------------------------------------------------------------------
// Generic bf16 GEMM: C[M,N] = A[M,K] @ Bt[N,K]^T + bias, optional ReLU,
// optional fp32 / bf16 outputs. 128x128 block, 8 waves, tileK = 32,
// double-buffered async global->LDS staging.
// ---------------------------------------------------------------------------
template <bool RELU, bool OUTF32, bool OUTBF>
__global__ __launch_bounds__(256)
void gemm_bf16_kernel(const bf16_t* __restrict__ A, const bf16_t* __restrict__ Bt,
                      const float* __restrict__ bias, float* __restrict__ Cf,
                      bf16_t* __restrict__ Cb, int N, int K) {
  const int bm = blockIdx.y * 128, bn = blockIdx.x * 128;
  const int tid = threadIdx.x, lane = tid & 31, wave = tid >> 5;
  const int wm = (wave & 3) * 32;    // 4 waves along M, 32 rows each
  const int wn = (wave >> 2) * 64;   // 2 waves along N, 64 cols each

  __shared__ __align__(16) bf16_t As[2][128][40];
  __shared__ __align__(16) bf16_t Bs[2][128][40];

  const v8f vz = {0.f, 0.f, 0.f, 0.f, 0.f, 0.f, 0.f, 0.f};
  v8f acc[2][4];
#pragma unroll
  for (int mi = 0; mi < 2; ++mi)
#pragma unroll
    for (int ni = 0; ni < 4; ++ni) acc[mi][ni] = vz;

  const int lrow = tid >> 1, half = tid & 1;   // 2 threads per 32-wide row
  const bf16_t* Ag = A  + (size_t)(bm + lrow) * K + half * 16;
  const bf16_t* Bg = Bt + (size_t)(bn + lrow) * K + half * 16;

  // Issue async copies of one 128x32 A tile + 128x32 B tile (4 x b128/thread).
  auto issue_tile = [&](int k0, int buf) {
    async_copy_b128(&As[buf][lrow][half * 16],     Ag + k0);
    async_copy_b128(&As[buf][lrow][half * 16 + 8], Ag + k0 + 8);
    async_copy_b128(&Bs[buf][lrow][half * 16],     Bg + k0);
    async_copy_b128(&Bs[buf][lrow][half * 16 + 8], Bg + k0 + 8);
  };

  issue_tile(0, 0);
  for (int k0 = 0; k0 < K; k0 += 32) {
    const int buf = (k0 >> 5) & 1;
    if (k0 + 32 < K) {
      issue_tile(k0 + 32, buf ^ 1);   // prefetch next tile into other buffer
      wait_asynccnt<4>();             // current tile's 4 copies have landed
    } else {
      wait_asynccnt<0>();
    }
    __syncthreads();                  // all waves' copies visible

    v16bf a0 = load_frag16(&As[buf][wm][0],      40, lane);
    v16bf a1 = load_frag16(&As[buf][wm + 16][0], 40, lane);
#pragma unroll
    for (int ni = 0; ni < 4; ++ni) {
      v16bf bfr = load_frag16(&Bs[buf][wn + ni * 16][0], 40, lane);
      acc[0][ni] = wmma_bf16(a0, bfr, acc[0][ni]);
      acc[1][ni] = wmma_bf16(a1, bfr, acc[1][ni]);
    }
    __syncthreads();                  // reads done before buf is re-filled
  }

  // C layout (ISA 7.12.2): VGPR r -> row r (lanes 0-15) / r+8 (lanes 16-31),
  // col = lane & 15.
  const int rofs = ((lane >> 4) & 1) * 8, cin = lane & 15;
#pragma unroll
  for (int mi = 0; mi < 2; ++mi)
#pragma unroll
    for (int ni = 0; ni < 4; ++ni) {
      const int col = bn + wn + ni * 16 + cin;
      const float bb = bias[col];
#pragma unroll
      for (int r = 0; r < 8; ++r) {
        float vv = acc[mi][ni][r] + bb;
        if (RELU) vv = fmaxf(vv, 0.0f);
        const size_t off = (size_t)(bm + wm + mi * 16 + r + rofs) * N + col;
        if (OUTF32) Cf[off] = vv;
        if (OUTBF)  Cb[off] = f2bf(vv);
      }
    }
}

// ---------------------------------------------------------------------------
// Weight prep: fp32 W[K][N] -> bf16 Wt[N][K]
// ---------------------------------------------------------------------------
__global__ __launch_bounds__(256)
void transpose_w_kernel(const float* __restrict__ W, bf16_t* __restrict__ Wt,
                        int K, int N) {
  const size_t idx = (size_t)blockIdx.x * 256 + threadIdx.x;
  const int k = (int)(idx / N), n = (int)(idx % N);
  Wt[(size_t)n * K + k] = f2bf(W[idx]);
}

// h = x + pos_emb (fp32 + bf16 copies)
__global__ __launch_bounds__(256)
void add_pos_kernel(const float* __restrict__ x, const float* __restrict__ pos,
                    float* __restrict__ h32, bf16_t* __restrict__ hbf) {
  const size_t idx = (size_t)blockIdx.x * 256 + threadIdx.x;
  const size_t sd = idx % ((size_t)kS * kD);
  const float v = x[idx] + pos[sd];
  h32[idx] = v;
  hbf[idx] = f2bf(v);
}

// LayerNorm over D=512, one wave per row; writes fp32 + bf16.
__global__ __launch_bounds__(256)
void ln_kernel(const float* __restrict__ in, const float* __restrict__ g,
               const float* __restrict__ bta, float* __restrict__ out32,
               bf16_t* __restrict__ outbf) {
  const int wave = threadIdx.x >> 5, lane = threadIdx.x & 31;
  const int row = blockIdx.x * 8 + wave;
  const float* p = in + (size_t)row * kD;
  float vals[16], s = 0.f, s2 = 0.f;
#pragma unroll
  for (int i = 0; i < 16; ++i) {
    const float v = p[lane + i * 32];
    vals[i] = v; s += v; s2 += v * v;
  }
#pragma unroll
  for (int m = 16; m; m >>= 1) {
    s  += __shfl_xor(s,  m, 32);
    s2 += __shfl_xor(s2, m, 32);
  }
  const float mu  = s * (1.0f / kD);
  const float var = s2 * (1.0f / kD) - mu * mu;
  const float inv = rsqrtf(var + 1e-5f);
#pragma unroll
  for (int i = 0; i < 16; ++i) {
    const int c = lane + i * 32;
    const float o = (vals[i] - mu) * inv * g[c] + bta[c];
    out32[(size_t)row * kD + c] = o;
    outbf[(size_t)row * kD + c] = f2bf(o);
  }
}

// LSH bucketing: bucket = argmax(concat(q@rot, -q@rot)) per (r,b,h,s).
__global__ __launch_bounds__(256)
void bucket_kernel(const float* __restrict__ qk, const float* __restrict__ rot,
                   int* __restrict__ buckets) {
  __shared__ float rotm[kDH * 32];   // 8 KB, one (r,h) slice
  const int tid = threadIdx.x;
  const int r = blockIdx.z / kH, hh = blockIdx.z % kH, b = blockIdx.y;
  const int s = blockIdx.x * 256 + tid;
  const float* rsrc = rot + (size_t)(r * kH + hh) * kDH * 32;
  for (int i = tid; i < kDH * 32; i += 256) rotm[i] = rsrc[i];
  __syncthreads();

  const float* q = qk + ((size_t)(b * kS + s)) * kD + hh * kDH;
  float dotv[32];
#pragma unroll
  for (int n_ = 0; n_ < 32; ++n_) dotv[n_] = 0.f;
#pragma unroll
  for (int db = 0; db < 4; ++db) {
    float qv[16];
#pragma unroll
    for (int j = 0; j < 16; ++j) qv[j] = q[db * 16 + j];
#pragma unroll
    for (int j = 0; j < 16; ++j)
#pragma unroll
      for (int n_ = 0; n_ < 32; ++n_)
        dotv[n_] = fmaf(qv[j], rotm[(db * 16 + j) * 32 + n_], dotv[n_]);
  }
  float best = -3.0e38f; int bi = 0;
#pragma unroll
  for (int n_ = 0; n_ < 32; ++n_)
    if (dotv[n_] > best) { best = dotv[n_]; bi = n_; }
#pragma unroll
  for (int n_ = 0; n_ < 32; ++n_)
    if (-dotv[n_] > best) { best = -dotv[n_]; bi = n_ + 32; }
  const int rbh = (r * kB + b) * kH + hh;
  buckets[(size_t)rbh * kS + s] = bi;
}

// Stable counting sort of 4096 tokens into 64 buckets, one WG per (r,b,h).
__global__ __launch_bounds__(128)
void sort_kernel(const int* __restrict__ buckets, int* __restrict__ sticker,
                 int* __restrict__ undo) {
  __shared__ uint32_t cnt[64][128];   // 32 KB
  __shared__ uint32_t ctot[128];
  const int tid = threadIdx.x;
  const size_t base = (size_t)blockIdx.x * kS;
#pragma unroll 8
  for (int j = 0; j < 64; ++j) cnt[j][tid] = 0;
  __syncthreads();
  for (int e = 0; e < 32; ++e) {
    const int bk = buckets[base + tid * 32 + e];
    cnt[bk][tid]++;
  }
  __syncthreads();
  {  // chunked exclusive prefix (bucket-major, tid-minor flattening)
    const int bkrow = tid >> 1, c0 = (tid & 1) * 64;
    uint32_t run = 0;
    for (int e = 0; e < 64; ++e) {
      const uint32_t t = cnt[bkrow][c0 + e];
      cnt[bkrow][c0 + e] = run;
      run += t;
    }
    ctot[tid] = run;
  }
  __syncthreads();
  if (tid == 0) {
    uint32_t run = 0;
    for (int t2 = 0; t2 < 128; ++t2) {
      const uint32_t t = ctot[t2]; ctot[t2] = run; run += t;
    }
  }
  __syncthreads();
  {
    const int bkrow = tid >> 1, c0 = (tid & 1) * 64;
    const uint32_t add = ctot[tid];
    for (int e = 0; e < 64; ++e) cnt[bkrow][c0 + e] += add;
  }
  __syncthreads();
  for (int e = 0; e < 32; ++e) {
    const int s = tid * 32 + e;
    const int bk = buckets[base + s];
    const uint32_t off = cnt[bk][tid]++;
    sticker[base + off] = s;
    undo[base + s] = (int)off;
  }
}

// Gather sorted QK (row-major bf16), V (transposed [dh][s] bf16), buckets.
__global__ __launch_bounds__(256)
void gather_kernel(const float* __restrict__ qk, const float* __restrict__ v,
                   const int* __restrict__ sticker, const int* __restrict__ buckets,
                   bf16_t* __restrict__ sqk_s, bf16_t* __restrict__ sv_t,
                   int* __restrict__ sbuck) {
  const int gi = blockIdx.x * 256 + threadIdx.x;
  const int rbh = gi >> 12, i = gi & (kS - 1);
  const size_t base = (size_t)rbh * kS;
  const int j = sticker[base + i];
  const int hh = rbh % kH, b = (rbh / kH) % kB;
  const float* qs = qk + ((size_t)(b * kS + j)) * kD + hh * kDH;
  const float* vs = v  + ((size_t)(b * kS + j)) * kD + hh * kDH;
  bf16_t* qd = sqk_s + (base + i) * kDH;
#pragma unroll
  for (int d = 0; d < kDH; ++d) {
    qd[d] = f2bf(qs[d]);
    sv_t[((size_t)rbh * kDH + d) * kS + i] = f2bf(vs[d]);
  }
  sbuck[base + i] = buckets[base + j];
}

// Chunked LSH attention: one WG per (rbh, chunk). Scores + softmax + PV.
__global__ __launch_bounds__(256)
void attn_kernel(const bf16_t* __restrict__ sqk_s, const bf16_t* __restrict__ sv_t,
                 const int* __restrict__ sbuck, float* __restrict__ o_sorted) {
  __shared__ __align__(16) bf16_t Kf[128][72];   // prev||cur K rows (Q = rows 64..127)
  __shared__ __align__(16) float  Sc[64][132];   // scores fp32, later aliased bf16 probs
  __shared__ int Bkt[128];
  const int tid = threadIdx.x, lane = tid & 31, wave = tid >> 5;
  const int rbh = blockIdx.x >> 6, n = blockIdx.x & (kNB - 1);
  const int prev = (n + kNB - 1) & (kNB - 1);
  const size_t base = (size_t)rbh * kS;

  {  // stage K tile (async global->LDS) + buckets
    const int row = tid >> 1, half = tid & 1;
    const int isrc = (row < 64) ? (prev * 64 + row) : (n * 64 + (row - 64));
    const bf16_t* src = sqk_s + (base + isrc) * kDH + half * 32;
    bf16_t* dst = &Kf[row][half * 32];
#pragma unroll
    for (int j2 = 0; j2 < 4; ++j2)
      async_copy_b128(dst + j2 * 8, src + j2 * 8);
    if (tid < 128) {
      const int i2 = (tid < 64) ? (prev * 64 + tid) : (n * 64 + (tid - 64));
      Bkt[tid] = sbuck[base + i2];
    }
    wait_asynccnt<0>();
  }
  __syncthreads();

  {  // scores = Q @ K^T * (1/sqrt(64)); 4x8 tiles over 8 waves
    const int mi = wave & 3, ng = (wave >> 2) * 4;
    const v8f vz = {0.f, 0.f, 0.f, 0.f, 0.f, 0.f, 0.f, 0.f};
    v8f sacc[4];
#pragma unroll
    for (int t = 0; t < 4; ++t) sacc[t] = vz;
#pragma unroll
    for (int ks = 0; ks < 2; ++ks) {
      const int k0 = ks * 32;
      v16bf aq = load_frag16(&Kf[64 + mi * 16][k0], 72, lane);
#pragma unroll
      for (int t = 0; t < 4; ++t) {
        v16bf bk16 = load_frag16(&Kf[(ng + t) * 16][k0], 72, lane);
        sacc[t] = wmma_bf16(aq, bk16, sacc[t]);
      }
    }
    const int rofs = ((lane >> 4) & 1) * 8, cin = lane & 15;
#pragma unroll
    for (int t = 0; t < 4; ++t)
#pragma unroll
      for (int r = 0; r < 8; ++r)
        Sc[mi * 16 + r + rofs][(ng + t) * 16 + cin] = sacc[t][r] * 0.125f;
  }
  __syncthreads();

  if (tid < 64) {  // masked softmax per row; rewrite probs bf16 in place
    const int row = tid;
    const int bq = Bkt[64 + row];
    float mx = -3.0e38f;
    for (int j2 = 0; j2 < 128; ++j2) {
      const float sv = (Bkt[j2] == bq) ? Sc[row][j2] : -1.0e9f;
      Sc[row][j2] = sv;
      mx = fmaxf(mx, sv);
    }
    float sm = 0.f;
    for (int j2 = 0; j2 < 128; ++j2) {
      const float e = __expf(Sc[row][j2] - mx);
      Sc[row][j2] = e;
      sm += e;
    }
    const float is = 1.0f / sm;
    bf16_t* arow = (bf16_t*)&Sc[row][0];
    for (int j2 = 0; j2 < 128; ++j2) arow[j2] = f2bf(Sc[row][j2] * is);
  }
  __syncthreads();

  {  // O = P @ V; B-fragments streamed from global transposed V
    const int mi = wave >> 1, njb = (wave & 1) * 2;
    const int cin = lane & 15, ksel = (lane >> 4) & 1, rofs = ksel * 8;
    const bf16_t* attnb = (const bf16_t*)&Sc[0][0];   // stride 264 bf16/row
    const v8f vz = {0.f, 0.f, 0.f, 0.f, 0.f, 0.f, 0.f, 0.f};
    v8f oacc[2];
    oacc[0] = vz; oacc[1] = vz;
#pragma unroll
    for (int ks = 0; ks < 4; ++ks) {
      const int k0 = ks * 32;
      v16bf aa = load_frag16(attnb + mi * 16 * 264 + k0, 264, lane);
      const int kk = k0 + ksel * 8;
      const int i0 = (kk < 64) ? (prev * 64 + kk) : (n * 64 + (kk - 64));
#pragma unroll
      for (int t = 0; t < 2; ++t) {
        const int d = (njb + t) * 16 + cin;
        const bf16_t* vp = sv_t + ((size_t)rbh * kDH + d) * kS + i0;
        v8bf lo = *(const v8bf*)vp;
        v8bf hi = *(const v8bf*)(vp + 16);
        v16bf bb = __builtin_shufflevector(lo, hi,
                     0,1,2,3,4,5,6,7,8,9,10,11,12,13,14,15);
        oacc[t] = wmma_bf16(aa, bb, oacc[t]);
      }
    }
#pragma unroll
    for (int t = 0; t < 2; ++t)
#pragma unroll
      for (int r = 0; r < 8; ++r) {
        const int ii = n * 64 + mi * 16 + r + rofs;
        const int d = (njb + t) * 16 + cin;
        o_sorted[(base + ii) * kDH + d] = oacc[t][r];
      }
  }
}

// Unsort + average over R rounds + head-merge -> bf16 [B,S,D]
__global__ __launch_bounds__(256)
void unsort_kernel(const float* __restrict__ o_sorted, const int* __restrict__ undo,
                   bf16_t* __restrict__ o_bf) {
  const int gi = blockIdx.x * 256 + threadIdx.x;
  const int hh = gi % kH;
  const int t2 = gi / kH;
  const int s = t2 % kS, b = t2 / kS;
  const int rbh0 = (0 * kB + b) * kH + hh;
  const int rbh1 = (1 * kB + b) * kH + hh;
  const int u0 = undo[(size_t)rbh0 * kS + s];
  const int u1 = undo[(size_t)rbh1 * kS + s];
  const float* p0 = o_sorted + ((size_t)rbh0 * kS + u0) * kDH;
  const float* p1 = o_sorted + ((size_t)rbh1 * kS + u1) * kDH;
  bf16_t* dst = o_bf + ((size_t)(b * kS + s)) * kD + hh * kDH;
#pragma unroll
  for (int d = 0; d < kDH; ++d) dst[d] = f2bf(0.5f * (p0[d] + p1[d]));
}

// ---------------------------------------------------------------------------
// Host orchestration
// ---------------------------------------------------------------------------
extern "C" void kernel_launch(void* const* d_in, const int* in_sizes, int n_in,
                              void* d_out, int out_size, void* d_ws, size_t ws_size,
                              hipStream_t stream) {
  (void)in_sizes; (void)n_in; (void)out_size; (void)ws_size;

  const float* x    = (const float*)d_in[0];
  const float* pos  = (const float*)d_in[1];
  const float* Wqk  = (const float*)d_in[2];
  const float* bqk  = (const float*)d_in[3];
  const float* Wv   = (const float*)d_in[4];
  const float* bv   = (const float*)d_in[5];
  const float* Wo   = (const float*)d_in[6];
  const float* bo   = (const float*)d_in[7];
  const float* rot  = (const float*)d_in[8];
  const float* ln_g = (const float*)d_in[9];
  const float* ln_b = (const float*)d_in[10];
  const float* W1   = (const float*)d_in[11];
  const float* b1   = (const float*)d_in[12];
  const float* W2   = (const float*)d_in[13];
  const float* b2   = (const float*)d_in[14];

  char* wp = (char*)d_ws;
  auto alloc = [&](size_t bytes) -> void* {
    void* p = (void*)wp;
    wp += (bytes + 255) & ~(size_t)255;
    return p;
  };
  float*  h32      = (float*) alloc((size_t)kM * kD * 4);
  bf16_t* hbf      = (bf16_t*)alloc((size_t)kM * kD * 2);
  float*  qk32     = (float*) alloc((size_t)kM * kD * 4);
  float*  v32      = (float*) alloc((size_t)kM * kD * 4);
  bf16_t* o_bf     = (bf16_t*)alloc((size_t)kM * kD * 2);
  bf16_t* mid_bf   = (bf16_t*)alloc((size_t)kM * kDFF * 2);
  bf16_t* WqkT     = (bf16_t*)alloc((size_t)kD * kD * 2);
  bf16_t* WvT      = (bf16_t*)alloc((size_t)kD * kD * 2);
  bf16_t* WoT      = (bf16_t*)alloc((size_t)kD * kD * 2);
  bf16_t* W1T      = (bf16_t*)alloc((size_t)kD * kDFF * 2);
  bf16_t* W2T      = (bf16_t*)alloc((size_t)kDFF * kD * 2);
  int*    buckets  = (int*)   alloc((size_t)kRBH * kS * 4);
  int*    sticker  = (int*)   alloc((size_t)kRBH * kS * 4);
  int*    undo     = (int*)   alloc((size_t)kRBH * kS * 4);
  int*    sbuck    = (int*)   alloc((size_t)kRBH * kS * 4);
  bf16_t* sqk_s    = (bf16_t*)alloc((size_t)kRBH * kS * kDH * 2);
  bf16_t* sv_t     = (bf16_t*)alloc((size_t)kRBH * kS * kDH * 2);
  float*  o_sorted = (float*) alloc((size_t)kRBH * kS * kDH * 4);

  // One-time weight prep (bf16, transposed to [N][K]).
  transpose_w_kernel<<<(kD * kD) / 256, 256, 0, stream>>>(Wqk, WqkT, kD, kD);
  transpose_w_kernel<<<(kD * kD) / 256, 256, 0, stream>>>(Wv,  WvT,  kD, kD);
  transpose_w_kernel<<<(kD * kD) / 256, 256, 0, stream>>>(Wo,  WoT,  kD, kD);
  transpose_w_kernel<<<(kD * kDFF) / 256, 256, 0, stream>>>(W1, W1T, kD, kDFF);
  transpose_w_kernel<<<(kDFF * kD) / 256, 256, 0, stream>>>(W2, W2T, kDFF, kD);

  add_pos_kernel<<<(kM * kD) / 256, 256, 0, stream>>>(x, pos, h32, hbf);

  const dim3 gProj(kD / 128, kM / 128);     // 512-wide GEMMs
  const dim3 gFF1(kDFF / 128, kM / 128);    // FFN up

  for (int l = 0; l < 4; ++l) {
    gemm_bf16_kernel<false, true, false><<<gProj, 256, 0, stream>>>(
        hbf, WqkT, bqk, qk32, nullptr, kD, kD);
    gemm_bf16_kernel<false, true, false><<<gProj, 256, 0, stream>>>(
        hbf, WvT, bv, v32, nullptr, kD, kD);

    bucket_kernel<<<dim3(kS / 256, kB, kR * kH), 256, 0, stream>>>(qk32, rot, buckets);
    sort_kernel<<<kRBH, 128, 0, stream>>>(buckets, sticker, undo);
    gather_kernel<<<(kRBH * kS) / 256, 256, 0, stream>>>(
        qk32, v32, sticker, buckets, sqk_s, sv_t, sbuck);
    attn_kernel<<<kRBH * kNB, 256, 0, stream>>>(sqk_s, sv_t, sbuck, o_sorted);
    unsort_kernel<<<(kB * kS * kH) / 256, 256, 0, stream>>>(o_sorted, undo, o_bf);

    gemm_bf16_kernel<false, true, false><<<gProj, 256, 0, stream>>>(
        o_bf, WoT, bo, h32, nullptr, kD, kD);
    ln_kernel<<<kM / 8, 256, 0, stream>>>(h32, ln_g, ln_b, h32, hbf);

    gemm_bf16_kernel<true, false, true><<<gFF1, 256, 0, stream>>>(
        hbf, W1T, b1, nullptr, mid_bf, kDFF, kD);
    gemm_bf16_kernel<false, true, false><<<gProj, 256, 0, stream>>>(
        mid_bf, W2T, b2, h32, nullptr, kD, kDFF);

    float* lnout = (l == 3) ? (float*)d_out : h32;
    ln_kernel<<<kM / 8, 256, 0, stream>>>(h32, ln_g, ln_b, lnout, hbf);
  }
}
